// Attention_85650237817127
// MI455X (gfx1250) — matrix operand order repaired and checked by reading, not compiled, not run on hardware
//
#include <hip/hip_runtime.h>

typedef __bf16 bf16_t;
typedef bf16_t v16bf __attribute__((ext_vector_type(16)));
typedef bf16_t v8bf  __attribute__((ext_vector_type(8)));
typedef float  v8f   __attribute__((ext_vector_type(8)));

#define N_EMBD   1024
#define N_HEAD   16
#define HEAD_DIM 64
#define SEQ      2048
#define BATCH    4

__device__ __forceinline__ bf16_t f2bf(float f) {
  union { float f; unsigned u; } a; a.f = f;
  unsigned r = a.u + 0x7FFFu + ((a.u >> 16) & 1u);   // round-to-nearest-even
  union { unsigned short s; bf16_t b; } o; o.s = (unsigned short)(r >> 16);
  return o.b;
}

__device__ __forceinline__ v8f wmma_bf16(v16bf a, v16bf b, v8f c) {
  // (neg_a, A, neg_b, B, c_mod, C, reuse_a, reuse_b)
  return __builtin_amdgcn_wmma_f32_16x16x32_bf16(false, a, false, b, (short)0, c, false, false);
}

// A-matrix 16x32 bf16 fragment as loaded here: lane m=l%16, g=l/16,
// element i -> k = 16g + i (contiguous, matches load_b_frag).
__device__ __forceinline__ v16bf load_a_frag(const bf16_t* row, int g) {
  v8bf lo = *(const v8bf*)(row + 16 * g);
  v8bf hi = *(const v8bf*)(row + 16 * g + 8);
  v16bf a;
#pragma unroll
  for (int i = 0; i < 8; ++i) { a[i] = lo[i]; a[8 + i] = hi[i]; }
  return a;
}

// B-matrix 32x16 bf16 fragment: lane n=l%16, g=l/16, element i -> k = 16g + i
// (one contiguous 32-byte run when source is k-major).
__device__ __forceinline__ v16bf load_b_frag(const bf16_t* col, int g) {
  return *(const v16bf*)(col + 16 * g);
}

// ---- gfx1250 async global->LDS copy (16B per lane), ASYNCcnt-tracked ----
#if __has_builtin(__builtin_amdgcn_global_load_async_to_lds_b128)
#define HAVE_ASYNC_LDS 1
typedef int v4i_raw __attribute__((vector_size(16)));                 // matches builtin proto
typedef __attribute__((address_space(1))) v4i_raw gv4i_t;             // global
typedef __attribute__((address_space(3))) v4i_raw lv4i_t;             // LDS
__device__ __forceinline__ void cp16(bf16_t* lds, const bf16_t* gsrc) {
  // generic LDS address -> low 32 bits are the LDS byte offset (ISA aperture rule)
  __builtin_amdgcn_global_load_async_to_lds_b128(
      (gv4i_t*)(uintptr_t)gsrc,
      (lv4i_t*)(uintptr_t)(unsigned)(uintptr_t)lds,
      0, 0);
}
template <int N>
__device__ __forceinline__ void wait_async() {
#if __has_builtin(__builtin_amdgcn_s_wait_asynccnt)
  __builtin_amdgcn_s_wait_asynccnt(N);
#else
  asm volatile("s_wait_asynccnt %0" ::"i"(N) : "memory");
#endif
}
#else
__device__ __forceinline__ void cp16(bf16_t* lds, const bf16_t* gsrc) {
  *(v8bf*)lds = *(const v8bf*)gsrc;  // sync fallback
}
template <int N> __device__ __forceinline__ void wait_async() {}
#endif

// ---------------- conversion kernels ----------------
__global__ void __launch_bounds__(256)
k_f32_to_bf16(const float* __restrict__ in, bf16_t* __restrict__ out, int n) {
  int i = blockIdx.x * blockDim.x + threadIdx.x;
  if (i < n) out[i] = f2bf(in[i]);
}

// in: [K][N] f32 row-major  ->  out: [N][K] bf16 (k-major per output row)
__global__ void __launch_bounds__(256)
k_transpose_bf16(const float* __restrict__ in, bf16_t* __restrict__ out, int K, int N) {
  int idx = blockIdx.x * blockDim.x + threadIdx.x;
  if (idx < K * N) {
    int nrow = idx / K;
    int k    = idx % K;
    out[idx] = f2bf(in[(size_t)k * N + nrow]);
  }
}

// ---------------- QKV projection GEMM ----------------
// X[8192,1024]bf16 @ WattnT[3072,1024]bf16 (+bias f32) -> scatter Q,K [B,H,S,D], V [B,H,D,S]
__global__ void __launch_bounds__(256)
k_qkv_gemm(const bf16_t* __restrict__ X, const bf16_t* __restrict__ Wt,
           const float* __restrict__ bias,
           bf16_t* __restrict__ Q, bf16_t* __restrict__ Ko, bf16_t* __restrict__ Vt) {
  const int lane = threadIdx.x & 31;
  const int wave = threadIdx.x >> 5;
  const int g    = lane >> 4;
  const int nl   = lane & 15;
  const int row0 = blockIdx.x * 32 + (wave & 1) * 16;
  const int col0 = blockIdx.y * 256 + (wave >> 1) * 64;

  v8f acc[4];
#pragma unroll
  for (int t = 0; t < 4; ++t)
#pragma unroll
    for (int r = 0; r < 8; ++r) acc[t][r] = 0.f;

  const bf16_t* xrow = X + (size_t)(row0 + nl) * N_EMBD;
  for (int kb = 0; kb < N_EMBD; kb += 32) {
    v16bf a = load_a_frag(xrow + kb, g);
#pragma unroll
    for (int t = 0; t < 4; ++t) {
      const bf16_t* wcol = Wt + (size_t)(col0 + t * 16 + nl) * N_EMBD + kb;
      acc[t] = wmma_bf16(a, load_b_frag(wcol, g), acc[t]);
    }
  }

#pragma unroll
  for (int t = 0; t < 4; ++t) {
    int col = col0 + t * 16 + nl;
    float bv = bias[col];
    int which = col >> 10;      // 0=q 1=k 2=v (uniform per block)
    int e = col & 1023;
    int h = e >> 6, d = e & 63;
#pragma unroll
    for (int r = 0; r < 8; ++r) {
      int row = row0 + r + 8 * g;
      int b = row >> 11, s = row & 2047;
      bf16_t v = f2bf(acc[t][r] + bv);
      size_t bh = (size_t)(b * N_HEAD + h);
      if (which == 0)      Q [(bh * SEQ + s) * HEAD_DIM + d] = v;
      else if (which == 1) Ko[(bh * SEQ + s) * HEAD_DIM + d] = v;
      else                 Vt[(bh * HEAD_DIM + d) * SEQ + s] = v;
    }
  }
}

// ---------------- flash attention (causal, online softmax) ----------------
// grid.x = B*H, grid.y = S/128; 8 waves/block, each wave owns one 16-query tile.
// K (32x64) and V^T (64x32) chunks are staged block-cooperatively into LDS with
// async global->LDS DMA, double-buffered: chunk c+1 streams in while c computes.
__global__ void __launch_bounds__(256)
k_flash(const bf16_t* __restrict__ Q, const bf16_t* __restrict__ Kc,
        const bf16_t* __restrict__ Vt, bf16_t* __restrict__ O) {
  const int tid  = threadIdx.x;
  const int lane = tid & 31;
  const int wave = tid >> 5;
  const int g    = lane >> 4;
  const int nl   = lane & 15;
  const int bh   = blockIdx.x;
  const int b    = bh >> 4, h = bh & 15;
  const int q0b  = blockIdx.y * 128;
  const int q0   = q0b + wave * 16;

  const bf16_t* Qp = Q  + (size_t)bh * SEQ * HEAD_DIM;
  const bf16_t* Kp = Kc + (size_t)bh * SEQ * HEAD_DIM;
  const bf16_t* Vp = Vt + (size_t)bh * HEAD_DIM * SEQ;

  __shared__ __align__(64) bf16_t sK[2][32][HEAD_DIM];  // keys x dims   (2x4KB)
  __shared__ __align__(64) bf16_t sV[2][HEAD_DIM][32];  // dims x keys   (2x4KB)
  __shared__ __align__(64) float  sS[8][16][32];
  __shared__ __align__(64) bf16_t sP[8][16][32];
  __shared__ float sMax[8][16], sSum[8][16], sScale[8][16];

  const bf16_t* qrow = Qp + (size_t)(q0 + nl) * HEAD_DIM;
  v16bf a0 = load_a_frag(qrow, g);        // dims 0..31
  v16bf a1 = load_a_frag(qrow + 32, g);   // dims 32..63

  v8f o[4];
#pragma unroll
  for (int t = 0; t < 4; ++t)
#pragma unroll
    for (int r = 0; r < 8; ++r) o[t][r] = 0.f;

  if (lane < 16) { sMax[wave][lane] = -3.0e38f; sSum[wave][lane] = 0.f; }

  // cooperative stage of one 32-key chunk: 256 lanes x 16B covers K and V
  auto stage = [&](int chunk, int bufi) {
    int kb = chunk * 32;
    cp16(&sK[bufi][tid >> 3][(tid & 7) * 8],
         Kp + (size_t)(kb + (tid >> 3)) * HEAD_DIM + (tid & 7) * 8);
    cp16(&sV[bufi][tid >> 2][(tid & 3) * 8],
         Vp + (size_t)(tid >> 2) * SEQ + kb + (tid & 3) * 8);
  };

  const int nchunks = (q0b + 128) / 32;   // causal upper bound for this block
  stage(0, 0);
  if (nchunks > 1) stage(1, 1);

  for (int c = 0; c < nchunks; ++c) {
    const int kb = c * 32;
    // chunk c complete (chunk c+1 may still be in flight), then make it
    // visible to every wave in the block
    if (c + 1 < nchunks) wait_async<2>(); else wait_async<0>();
    __syncthreads();

    const int bufi = c & 1;
    if (kb < q0 + 16) {                    // causal: this wave still has work
      v8f c0, c1;
#pragma unroll
      for (int r = 0; r < 8; ++r) { c0[r] = 0.f; c1[r] = 0.f; }

      const bf16_t* krow0 = &sK[bufi][nl][0];
      const bf16_t* krow1 = &sK[bufi][16 + nl][0];
      c0 = wmma_bf16(a0, load_b_frag(krow0, g), c0);
      c0 = wmma_bf16(a1, load_b_frag(krow0 + 32, g), c0);
      c1 = wmma_bf16(a0, load_b_frag(krow1, g), c1);
      c1 = wmma_bf16(a1, load_b_frag(krow1 + 32, g), c1);

      // scale 1/sqrt(64), GPT-2 causal bias (-10000 on masked), stage to LDS
#pragma unroll
      for (int r = 0; r < 8; ++r) {
        int row = r + 8 * g, qidx = q0 + row;
        sS[wave][row][nl]      = (kb + nl      <= qidx) ? c0[r] * 0.125f : -10000.0f;
        sS[wave][row][16 + nl] = (kb + 16 + nl <= qidx) ? c1[r] * 0.125f : -10000.0f;
      }

      // online softmax: lanes 0..15 own one row each (DS ops are wave-ordered)
      if (lane < 16) {
        int row = lane;
        float mOld = sMax[wave][row], mNew = mOld;
#pragma unroll
        for (int j = 0; j < 32; ++j) mNew = fmaxf(mNew, sS[wave][row][j]);
        float sc  = __expf(mOld - mNew);
        float sum = sSum[wave][row] * sc;
#pragma unroll
        for (int j = 0; j < 32; ++j) {
          float p = __expf(sS[wave][row][j] - mNew);
          sP[wave][row][j] = f2bf(p);
          sum += p;
        }
        sMax[wave][row] = mNew; sSum[wave][row] = sum; sScale[wave][row] = sc;
      }

      // rescale running O by per-row exp(mOld - mNew)
#pragma unroll
      for (int r = 0; r < 8; ++r) {
        float sc = sScale[wave][r + 8 * g];
#pragma unroll
        for (int t = 0; t < 4; ++t) o[t][r] *= sc;
      }

      // P (16x32) as A-fragment; V chunk as B-fragments, both from LDS
      v16bf ap = load_a_frag(&sP[wave][nl][0], g);
#pragma unroll
      for (int t = 0; t < 4; ++t)
        o[t] = wmma_bf16(ap, load_b_frag(&sV[bufi][t * 16 + nl][0], g), o[t]);
    }

    __syncthreads();                       // everyone done reading buf before reuse
    if (c + 2 < nchunks) stage(c + 2, bufi);
  }

  // finalize: divide by row sum, write merged-head [B,S,E] bf16
#pragma unroll
  for (int r = 0; r < 8; ++r) {
    int s = q0 + r + 8 * g;
    float inv = 1.0f / sSum[wave][r + 8 * g];
    size_t base = ((size_t)b * SEQ + s) * N_EMBD + h * HEAD_DIM;
#pragma unroll
    for (int t = 0; t < 4; ++t) O[base + t * 16 + nl] = f2bf(o[t][r] * inv);
  }
}

// ---------------- output projection GEMM ----------------
__global__ void __launch_bounds__(256)
k_proj_gemm(const bf16_t* __restrict__ A, const bf16_t* __restrict__ Wt,
            const float* __restrict__ bias, float* __restrict__ out) {
  const int lane = threadIdx.x & 31;
  const int wave = threadIdx.x >> 5;
  const int g    = lane >> 4;
  const int nl   = lane & 15;
  const int row0 = blockIdx.x * 32 + (wave & 1) * 16;
  const int col0 = blockIdx.y * 256 + (wave >> 1) * 64;

  v8f acc[4];
#pragma unroll
  for (int t = 0; t < 4; ++t)
#pragma unroll
    for (int r = 0; r < 8; ++r) acc[t][r] = 0.f;

  const bf16_t* arow = A + (size_t)(row0 + nl) * N_EMBD;
  for (int kb = 0; kb < N_EMBD; kb += 32) {
    v16bf a = load_a_frag(arow + kb, g);
#pragma unroll
    for (int t = 0; t < 4; ++t) {
      const bf16_t* wcol = Wt + (size_t)(col0 + t * 16 + nl) * N_EMBD + kb;
      acc[t] = wmma_bf16(a, load_b_frag(wcol, g), acc[t]);
    }
  }

#pragma unroll
  for (int t = 0; t < 4; ++t) {
    int col = col0 + t * 16 + nl;
    float bv = bias[col];
#pragma unroll
    for (int r = 0; r < 8; ++r) {
      int row = row0 + r + 8 * g;
      out[(size_t)row * N_EMBD + col] = acc[t][r] + bv;
    }
  }
}

extern "C" void kernel_launch(void* const* d_in, const int* in_sizes, int n_in,
                              void* d_out, int out_size, void* d_ws, size_t ws_size,
                              hipStream_t stream) {
  const float* x      = (const float*)d_in[0];  // [4,2048,1024]
  const float* w_attn = (const float*)d_in[1];  // [1024,3072]
  const float* b_attn = (const float*)d_in[2];  // [3072]
  const float* w_proj = (const float*)d_in[3];  // [1024,1024]
  const float* b_proj = (const float*)d_in[4];  // [1024]
  float* out = (float*)d_out;

  char* ws = (char*)d_ws;
  bf16_t* Xbf = (bf16_t*)(ws);                       // 16 MB  [8192,1024]
  bf16_t* WaT = (bf16_t*)(ws + 16777216);            //  6 MB  [3072,1024]
  bf16_t* WpT = (bf16_t*)(ws + 23068672);            //  2 MB  [1024,1024]
  bf16_t* Qbf = (bf16_t*)(ws + 25165824);            // 16 MB  [B,H,S,D]
  bf16_t* Kbf = (bf16_t*)(ws + 41943040);            // 16 MB  [B,H,S,D]
  bf16_t* Vtb = (bf16_t*)(ws + 58720256);            // 16 MB  [B,H,D,S]
  bf16_t* Obf = (bf16_t*)(ws + 75497472);            // 16 MB  [8192,1024]

  const int NX = BATCH * SEQ * N_EMBD;               // 8388608
  k_f32_to_bf16<<<(NX + 255) / 256, 256, 0, stream>>>(x, Xbf, NX);
  k_transpose_bf16<<<(N_EMBD * 3 * N_EMBD + 255) / 256, 256, 0, stream>>>(w_attn, WaT, N_EMBD, 3 * N_EMBD);
  k_transpose_bf16<<<(N_EMBD * N_EMBD + 255) / 256, 256, 0, stream>>>(w_proj, WpT, N_EMBD, N_EMBD);

  dim3 gq(256, 12);
  k_qkv_gemm<<<gq, 256, 0, stream>>>(Xbf, WaT, b_attn, Qbf, Kbf, Vtb);

  dim3 gf(BATCH * N_HEAD, SEQ / 128);
  k_flash<<<gf, 256, 0, stream>>>(Qbf, Kbf, Vtb, Obf);

  dim3 gp(256, 4);
  k_proj_gemm<<<gp, 256, 0, stream>>>(Obf, WpT, b_proj, out);
}
